// GCNII_1984274891523
// MI455X (gfx1250) — matrix operand order repaired and checked by reading, not compiled, not run on hardware
//
#include <hip/hip_runtime.h>
#include <hip/hip_bf16.h>
#include <math.h>

// GCNII on MI455X (gfx1250): fp16 WMMA for all large GEMMs, f32 accumulate.
// Dominant GEMM: a_norm[10000x10000] @ h[10000x256] per layer.
// a_norm stored once per call in fp16 *A-fragment* layout -> A loads are
// contiguous b128s, read exactly once per layer. h (5 MB) stored in
// *B-fragment* layout; each workgroup's 8 waves share B K-panels through a
// TRIPLE-buffered LDS pipeline fed by GLOBAL_LOAD_ASYNC_TO_LDS_B128
// (ASYNCcnt). B fragments are software-pipelined at depth 2 out of LDS so
// each WMMA waits on DS loads issued two tiles earlier (s_wait_dscnt <= 4),
// not the ones issued right before it.

typedef __attribute__((ext_vector_type(16))) _Float16 v16h;
typedef __attribute__((ext_vector_type(8)))  _Float16 v8h;
typedef __attribute__((ext_vector_type(8)))  float    v8f;

#define NN      10000
#define NFEAT   512
#define NHID    256
#define NCLASS  40
#define NLAYERS 8
#define MT      625    // M tiles of 16 (10000/16)
#define KTA     313    // K tiles of 32 for aggregation (10016 padded)
#define PANEL   8192   // halves per 32x256 B K-panel
#define USE_ASYNC_LDS 1

// ---------- WMMA fragment element-offset helpers (contiguous k order: k = 16*(lane/16) + slot) ----------
__device__ __forceinline__ size_t afrag_idx(int m, int k, int ktiles) {
  int mt = m >> 4, r = m & 15;
  int kt = k >> 5, c = k & 31;
  int lane = r + 16 * (c >> 4);
  int slot = c & 15;
  return ((size_t)(mt * ktiles + kt) * 512) + (size_t)(lane * 16 + slot);
}
__device__ __forceinline__ size_t bfrag_idx(int k, int n) {
  int kt = k >> 5, kc = k & 31;
  int nt = n >> 4, nc = n & 15;
  int lane = nc + 16 * (kc >> 4);
  int slot = kc & 15;
  return ((size_t)(kt * 16 + nt) * 512) + (size_t)(lane * 16 + slot);
}

// ---------- async global->LDS panel copy (16KB per panel, 4 b128 per thread) ----------
__device__ __forceinline__ void copy_panel(const _Float16* __restrict__ gsrc,
                                           _Float16* lds_dst, int tid) {
#pragma unroll
  for (int i = 0; i < 4; ++i) {
    const _Float16* g = gsrc + (size_t)(i * 256 + tid) * 8;
    _Float16* l = lds_dst + (size_t)(i * 256 + tid) * 8;
#if USE_ASYNC_LDS
    unsigned loff =
        (unsigned)(unsigned long long)(__attribute__((address_space(3))) _Float16*)l;
    unsigned long long ga = (unsigned long long)g;
    asm volatile("global_load_async_to_lds_b128 %0, %1, off"
                 :: "v"(loff), "v"(ga) : "memory");
#else
    *(v8h*)l = *(const v8h*)g;
#endif
  }
}
__device__ __forceinline__ void wait_async_le4() {
#if USE_ASYNC_LDS
  asm volatile("s_wait_asynccnt 0x4" ::: "memory");
#endif
}

// ---------- aggregation: hi = a_norm @ h ; support = 0.9*hi + 0.1*h0 ----------
__global__ void __launch_bounds__(256) k_agg(const _Float16* __restrict__ a16,
                                             const _Float16* __restrict__ hb,
                                             const float* __restrict__ h0f,
                                             float* __restrict__ supf,
                                             _Float16* __restrict__ supa) {
  __shared__ __align__(128) _Float16 sB[3][PANEL];
  const int tid = threadIdx.x;
  const int wave = tid >> 5, lane = tid & 31;
  const int mt = blockIdx.x * 8 + wave;
  const bool active = (mt < MT);        // wave-uniform

  // prologue: panels 0 and 1 in flight; wait for 0, keep 1 flying
  copy_panel(hb, &sB[0][0], tid);
  copy_panel(hb + PANEL, &sB[1][0], tid);
  wait_async_le4();
  __syncthreads();

  v8f c[16];
  v8f z = {0.f, 0.f, 0.f, 0.f, 0.f, 0.f, 0.f, 0.f};
#pragma unroll
  for (int i = 0; i < 16; ++i) c[i] = z;

  const _Float16* A = a16 + (size_t)mt * KTA * 512;
  v16h a = {};
  if (active) a = *(const v16h*)(A + lane * 16);

  int cur = 0;
  for (int kt = 0; kt < KTA; ++kt) {
    v16h a_next = {};
    if (active && (kt + 1 < KTA))
      a_next = *(const v16h*)(A + (size_t)(kt + 1) * 512 + lane * 16);
    if (active) {
      const _Float16* bp = &sB[cur][lane * 16];
      // depth-2 pipelined B fragments: wmma(nt) uses loads issued 2 tiles ago
      v16h b0 = *(const v16h*)(bp + 0 * 512);
      v16h b1 = *(const v16h*)(bp + 1 * 512);
#pragma unroll
      for (int nt = 0; nt < 16; ++nt) {
        v16h bn = {};
        if (nt + 2 < 16) bn = *(const v16h*)(bp + (nt + 2) * 512);
        c[nt] = __builtin_amdgcn_wmma_f32_16x16x32_f16(
            false, a, false, b0, (short)0, c[nt], false, false);
        b0 = b1;
        b1 = bn;
      }
    }
    // issue panel kt+2 into buffer (kt+2)%3 (its readers finished at barrier kt-1)
    int nxt2 = cur + 2; if (nxt2 >= 3) nxt2 -= 3;
    if (kt + 2 < KTA)
      copy_panel(hb + (size_t)(kt + 2) * PANEL, &sB[nxt2][0], tid);
    wait_async_le4();          // panel kt+1 landed (only kt+2's 4 ops may remain)
    __syncthreads();
    cur = cur + 1; if (cur >= 3) cur -= 3;
    a = a_next;
  }

  if (!active) return;
  int rbase = mt * 16 + 8 * (lane >> 4);
  int nc = lane & 15;
#pragma unroll
  for (int nt = 0; nt < 16; ++nt) {
#pragma unroll
    for (int v = 0; v < 8; ++v) {
      int row = rbase + v;
      int col = nt * 16 + nc;
      float sup = 0.9f * c[nt][v] + 0.1f * h0f[(size_t)row * NHID + col];
      supf[(size_t)row * NHID + col] = sup;
      supa[afrag_idx(row, col, 8)] = (_Float16)sup;   // A operand for GEMM2
    }
  }
}

// ---- pipelined small-GEMM core: 16 rows x 64 cols per wave, K-step prefetch ----
__device__ __forceinline__ void small_gemm(const _Float16* __restrict__ A,
                                           const _Float16* __restrict__ B,
                                           const int KT, int lane, v8f c[4]) {
  v16h a = *(const v16h*)(A + lane * 16);
  v16h b[4];
#pragma unroll
  for (int j = 0; j < 4; ++j) b[j] = *(const v16h*)(B + (size_t)j * 512 + lane * 16);
#pragma unroll 4
  for (int kt = 0; kt < KT; ++kt) {
    v16h an = {};
    v16h bn[4] = {};
    if (kt + 1 < KT) {
      an = *(const v16h*)(A + (size_t)(kt + 1) * 512 + lane * 16);
#pragma unroll
      for (int j = 0; j < 4; ++j)
        bn[j] = *(const v16h*)(B + (size_t)(kt + 1) * 8192 + (size_t)j * 512 + lane * 16);
    }
#pragma unroll
    for (int j = 0; j < 4; ++j)
      c[j] = __builtin_amdgcn_wmma_f32_16x16x32_f16(
          false, a, false, b[j], (short)0, c[j], false, false);
    a = an;
#pragma unroll
    for (int j = 0; j < 4; ++j) b[j] = bn[j];
  }
}

// ---------- h0 = relu(x @ w0 + b0); 16x64 per wave for occupancy ----------
__global__ void __launch_bounds__(256) k_h0(const _Float16* __restrict__ xa,
                                            const _Float16* __restrict__ w0b,
                                            const float* __restrict__ b0,
                                            float* __restrict__ h0f,
                                            _Float16* __restrict__ hb) {
  int w = blockIdx.x * 8 + (threadIdx.x >> 5);
  int lane = threadIdx.x & 31;
  int mt = w >> 2, ng = w & 3;          // 4 column groups of 64
  if (mt >= MT) return;                 // wave-uniform
  v8f c[4];
  v8f z = {0.f, 0.f, 0.f, 0.f, 0.f, 0.f, 0.f, 0.f};
#pragma unroll
  for (int i = 0; i < 4; ++i) c[i] = z;
  small_gemm(xa + (size_t)mt * 16 * 512, w0b + (size_t)(ng * 4) * 512, 16, lane, c);
  int rbase = mt * 16 + 8 * (lane >> 4);
  int nc = lane & 15;
#pragma unroll
  for (int j = 0; j < 4; ++j) {
#pragma unroll
    for (int v = 0; v < 8; ++v) {
      int row = rbase + v;
      int col = (ng * 4 + j) * 16 + nc;
      float o = fmaxf(c[j][v] + b0[col], 0.f);
      h0f[(size_t)row * NHID + col] = o;
      hb[bfrag_idx(row, col)] = (_Float16)o;
    }
  }
}

// ---------- h = relu(theta*(support@W) + (1-theta)*support); 16x64/wave ----------
__global__ void __launch_bounds__(256) k_gemm2(const _Float16* __restrict__ supa,
                                               const _Float16* __restrict__ wb,
                                               const float* __restrict__ supf,
                                               _Float16* __restrict__ hb,
                                               float* __restrict__ hfin,
                                               float theta) {
  int w = blockIdx.x * 8 + (threadIdx.x >> 5);
  int lane = threadIdx.x & 31;
  int mt = w >> 2, ng = w & 3;
  if (mt >= MT) return;
  v8f c[4];
  v8f z = {0.f, 0.f, 0.f, 0.f, 0.f, 0.f, 0.f, 0.f};
#pragma unroll
  for (int i = 0; i < 4; ++i) c[i] = z;
  small_gemm(supa + (size_t)mt * 8 * 512, wb + (size_t)(ng * 4) * 512, 8, lane, c);
  int rbase = mt * 16 + 8 * (lane >> 4);
  int nc = lane & 15;
#pragma unroll
  for (int j = 0; j < 4; ++j) {
#pragma unroll
    for (int v = 0; v < 8; ++v) {
      int row = rbase + v;
      int col = (ng * 4 + j) * 16 + nc;
      float s = supf[(size_t)row * NHID + col];
      float o = fmaxf(theta * c[j][v] + (1.0f - theta) * s, 0.f);
      hb[bfrag_idx(row, col)] = (_Float16)o;
      hfin[(size_t)row * NHID + col] = o;
    }
  }
}

// ---------- degree: dinv[i] = rsqrt(1 + sum_j adj[i][j]) ----------
__global__ void __launch_bounds__(256) k_rowsum(const float* __restrict__ adj,
                                                float* __restrict__ dinv) {
  __shared__ float s[256];
  int r = blockIdx.x;
  float acc = 0.f;
  for (int j = threadIdx.x; j < NN; j += 256) acc += adj[(size_t)r * NN + j];
  s[threadIdx.x] = acc;
  __syncthreads();
  for (int st = 128; st > 0; st >>= 1) {
    if (threadIdx.x < st) s[threadIdx.x] += s[threadIdx.x + st];
    __syncthreads();
  }
  if (threadIdx.x == 0) dinv[r] = rsqrtf(s[0] + 1.0f);
}

// ---------- a_norm -> fp16 A-fragment layout (K padded to 10016 with zeros) ----------
__global__ void __launch_bounds__(256) k_pack_adj(const float* __restrict__ adj,
                                                  const float* __restrict__ dinv,
                                                  _Float16* __restrict__ a16) {
  int blk = blockIdx.x;            // mt*KTA + kt
  int mt = blk / KTA, kt = blk % KTA;
  for (int e = threadIdx.x; e < 512; e += 256) {
    int r = e >> 5, c = e & 31;
    int row = mt * 16 + r;
    int col = kt * 32 + c;
    float v = 0.f;
    if (col < NN) {
      float a = adj[(size_t)row * NN + col];
      if (row == col) a += 1.0f;
      v = a * dinv[row] * dinv[col];
    }
    int lane = r + 16 * (c >> 4);
    int slot = c & 15;
    a16[(size_t)blk * 512 + lane * 16 + slot] = (_Float16)v;
  }
}

// ---------- x -> fp16 A-fragment (K=512, 16 K-tiles) ----------
__global__ void __launch_bounds__(256) k_pack_x(const float* __restrict__ x,
                                                _Float16* __restrict__ xa) {
  int blk = blockIdx.x;            // mt*16 + kt
  int mt = blk >> 4, kt = blk & 15;
  for (int e = threadIdx.x; e < 512; e += 256) {
    int r = e >> 5, c = e & 31;
    float v = x[(size_t)(mt * 16 + r) * NFEAT + kt * 32 + c];
    int lane = r + 16 * (c >> 4);
    int slot = c & 15;
    xa[(size_t)blk * 512 + lane * 16 + slot] = (_Float16)v;
  }
}

// ---------- w0[512x256] -> fp16 B-fragment ----------
__global__ void __launch_bounds__(256) k_pack_w0(const float* __restrict__ w0,
                                                 _Float16* __restrict__ wb) {
  int blk = blockIdx.x;            // kt*16 + nt
  int kt = blk >> 4, nt = blk & 15;
  for (int e = threadIdx.x; e < 512; e += 256) {
    int k = e >> 4, n = e & 15;
    float v = w0[(size_t)(kt * 32 + k) * NHID + nt * 16 + n];
    int lane = n + 16 * (k >> 4);
    int slot = k & 15;
    wb[(size_t)blk * 512 + lane * 16 + slot] = (_Float16)v;
  }
}

// ---------- conv_w[8][256][256] -> fp16 B-fragment per layer ----------
__global__ void __launch_bounds__(256) k_pack_convw(const float* __restrict__ cw,
                                                    _Float16* __restrict__ wb) {
  int blk = blockIdx.x;            // layer*128 + kt*16 + nt (kt<8, nt<16)
  int layer = blk >> 7, rem = blk & 127;
  int kt = rem >> 4, nt = rem & 15;
  const float* w = cw + (size_t)layer * NHID * NHID;
  for (int e = threadIdx.x; e < 512; e += 256) {
    int k = e >> 4, n = e & 15;
    float v = w[(size_t)(kt * 32 + k) * NHID + nt * 16 + n];
    int lane = n + 16 * (k >> 4);
    int slot = k & 15;
    wb[(size_t)blk * 512 + lane * 16 + slot] = (_Float16)v;
  }
}

// ---------- zero padded rows 10000..10015 in h B-fragment buffer ----------
__global__ void __launch_bounds__(256) k_pad_h(_Float16* __restrict__ hb) {
  for (int idx = threadIdx.x; idx < 4096; idx += 256) {
    int nt = idx >> 8, rem = idx & 255;
    int lane = 16 + (rem >> 4);
    int slot = rem & 15;
    hb[(size_t)(312 * 16 + nt) * 512 + lane * 16 + slot] = (_Float16)0.f;
  }
}

// ---------- logits = h @ w1 + b1 ; log_softmax (f32, per-row block) ----------
__global__ void __launch_bounds__(128) k_final(const float* __restrict__ hfin,
                                               const float* __restrict__ w1,
                                               const float* __restrict__ b1,
                                               float* __restrict__ out) {
  __shared__ float sh[NHID];
  __shared__ float lg[NCLASS];
  __shared__ float red[2];
  int r = blockIdx.x;
  for (int k = threadIdx.x; k < NHID; k += 128) sh[k] = hfin[(size_t)r * NHID + k];
  __syncthreads();
  if (threadIdx.x < NCLASS) {
    float acc = b1[threadIdx.x];
    for (int k = 0; k < NHID; ++k) acc += sh[k] * w1[k * NCLASS + threadIdx.x];
    lg[threadIdx.x] = acc;
  }
  __syncthreads();
  if (threadIdx.x == 0) {
    float m = lg[0];
    for (int i = 1; i < NCLASS; ++i) m = fmaxf(m, lg[i]);
    float s = 0.f;
    for (int i = 0; i < NCLASS; ++i) s += expf(lg[i] - m);
    red[0] = m;
    red[1] = logf(s);
  }
  __syncthreads();
  if (threadIdx.x < NCLASS)
    out[(size_t)r * NCLASS + threadIdx.x] = lg[threadIdx.x] - red[0] - red[1];
}

extern "C" void kernel_launch(void* const* d_in, const int* in_sizes, int n_in,
                              void* d_out, int out_size, void* d_ws, size_t ws_size,
                              hipStream_t stream) {
  const float* x   = (const float*)d_in[0];
  const float* adj = (const float*)d_in[1];
  const float* w0  = (const float*)d_in[2];
  const float* b0  = (const float*)d_in[3];
  const float* cw  = (const float*)d_in[4];
  const float* w1  = (const float*)d_in[5];
  const float* b1  = (const float*)d_in[6];
  float* out = (float*)d_out;

  // workspace layout (bytes, all 1KB-aligned); total ~254 MB
  char* ws = (char*)d_ws;
  const size_t OFF_DINV = 0;
  const size_t OFF_A16  = 40960;
  const size_t OFF_H    = OFF_A16 + 200320000ull;
  const size_t OFF_SUPA = OFF_H + 5128192ull;
  const size_t OFF_SUPF = OFF_SUPA + 5120000ull;
  const size_t OFF_H0F  = OFF_SUPF + 10240000ull;
  const size_t OFF_XA   = OFF_H0F + 10240000ull;
  const size_t OFF_W0B  = OFF_XA + 10240000ull;
  const size_t OFF_WB   = OFF_W0B + 262144ull;
  const size_t OFF_HFIN = OFF_WB + 2097152ull;

  float*    dinv = (float*)(ws + OFF_DINV);
  _Float16* a16  = (_Float16*)(ws + OFF_A16);
  _Float16* hb   = (_Float16*)(ws + OFF_H);
  _Float16* supa = (_Float16*)(ws + OFF_SUPA);
  float*    supf = (float*)(ws + OFF_SUPF);
  float*    h0f  = (float*)(ws + OFF_H0F);
  _Float16* xa   = (_Float16*)(ws + OFF_XA);
  _Float16* w0b  = (_Float16*)(ws + OFF_W0B);
  _Float16* wb   = (_Float16*)(ws + OFF_WB);
  float*    hfin = (float*)(ws + OFF_HFIN);

  // ---- preprocessing (every call; deterministic) ----
  k_rowsum<<<NN, 256, 0, stream>>>(adj, dinv);
  k_pack_adj<<<MT * KTA, 256, 0, stream>>>(adj, dinv, a16);
  k_pack_x<<<MT * 16, 256, 0, stream>>>(x, xa);
  k_pack_w0<<<256, 256, 0, stream>>>(w0, w0b);
  k_pack_convw<<<NLAYERS * 128, 256, 0, stream>>>(cw, wb);

  const int agg_blocks = (MT + 7) / 8;          // 8 waves, 16x256 each
  const int sg_blocks  = (MT * 4 + 7) / 8;      // 8 waves, 16x64 each
  k_h0<<<sg_blocks, 256, 0, stream>>>(xa, w0b, b0, h0f, hb);
  k_pad_h<<<1, 256, 0, stream>>>(hb);

  for (int l = 0; l < NLAYERS; ++l) {
    float theta = logf(0.5f / (float)(l + 1) + 1.0f);   // LAMDA = 0.5
    k_agg<<<agg_blocks, 256, 0, stream>>>(a16, hb, h0f, supf, supa);
    k_gemm2<<<sg_blocks, 256, 0, stream>>>(supa, wb + (size_t)l * 128 * 512,
                                           supf, hb, hfin, theta);
  }

  k_final<<<NN, 128, 0, stream>>>(hfin, w1, b1, out);
  (void)in_sizes; (void)n_in; (void)out_size; (void)ws_size;
}
